// Triplet_Loss_19061064859768
// MI455X (gfx1250) — compile-verified, matrix operand order checked by reading, not run on hardware
//
#include <hip/hip_runtime.h>

// ---------------------------------------------------------------------------
// Fused triplet loss for MI455X (gfx1250, wave32, WMMA).
//   D1[i,j] = || v_f[j]+eps - i_f[i] ||,  D2[i,j] = || i_f[j]+eps - v_f[i] ||
//   per-row: smax = max over same-label D, dmin = min over diff-label D
//   loss = sum relu(smax - dmin + 1.2) over both directions.
// Cross-term GEMM via v_wmma_f32_16x16x32_f16 (f32 accumulate); norm / eps
// terms exact f32. Double-buffered LDS with async global->LDS staging
// (global_load_async_to_lds_b128, ASYNCcnt) on gfx1250.
// ---------------------------------------------------------------------------

typedef __attribute__((ext_vector_type(16))) _Float16 v16h;
typedef __attribute__((ext_vector_type(8)))  float    v8f;

constexpr int   NB   = 4096;    // batch
constexpr int   ND   = 1024;    // feature dim
constexpr float FEPS = 1e-6f;
constexpr float FXI  = 1.2f;

constexpr int BM = 256, BN = 128, BK = 32;   // block tile; 8 waves of 64x64
constexpr int LROW = BK + 8;                 // padded LDS row (halves): 80B stride

#if defined(__gfx1250__) && __has_builtin(__builtin_amdgcn_global_load_async_to_lds_b128)
#define TL_ASYNC 1
// builtin prototype (from hipcc diagnostic): (v4i AS1*, v4i AS3*, Imm int, Imm int)
typedef int tl_v4i __attribute__((vector_size(16)));
typedef __attribute__((address_space(1))) tl_v4i* tl_gvp;   // global v4i*
typedef __attribute__((address_space(3))) tl_v4i* tl_lvp;   // LDS v4i*
#else
#define TL_ASYNC 0
#endif

// ---------------------------------------------------------------------------
__global__ void tl_init_kernel(unsigned* __restrict__ smax,
                               unsigned* __restrict__ dmin, int n) {
  int i = blockIdx.x * blockDim.x + threadIdx.x;
  if (i < n) { smax[i] = 0u; dmin[i] = 0x7f800000u; /* +inf */ }
}

// f32 -> f16 copies + per-row sum(x^2), sum(x)
__global__ void tl_prep_kernel(const float* __restrict__ i_f,
                               const float* __restrict__ v_f,
                               _Float16* __restrict__ h0,  // f16(i_f)
                               _Float16* __restrict__ h1,  // f16(v_f)
                               float* __restrict__ nsq,    // [2][NB]
                               float* __restrict__ neps,   // [2][NB]  ||x+eps||^2
                               float* __restrict__ rsum) { // [2][NB]  sum(x)
  const int t = blockIdx.y;           // 0 = i_f, 1 = v_f
  const int r = blockIdx.x;
  const float* src = (t == 0) ? i_f : v_f;
  _Float16*    dst = (t == 0) ? h0  : h1;

  float ss = 0.f, s = 0.f;
#pragma unroll
  for (int k = 0; k < ND / 256; ++k) {
    int c = threadIdx.x + k * 256;
    float x = src[(size_t)r * ND + c];
    dst[(size_t)r * ND + c] = (_Float16)x;
    ss += x * x;
    s  += x;
  }
  __shared__ float r0[256], r1[256];
  r0[threadIdx.x] = ss; r1[threadIdx.x] = s;
  __syncthreads();
  for (int o = 128; o > 0; o >>= 1) {
    if (threadIdx.x < (unsigned)o) {
      r0[threadIdx.x] += r0[threadIdx.x + o];
      r1[threadIdx.x] += r1[threadIdx.x + o];
    }
    __syncthreads();
  }
  if (threadIdx.x == 0) {
    nsq [t * NB + r] = r0[0];
    neps[t * NB + r] = r0[0] + 2.f * FEPS * r1[0] + (float)ND * FEPS * FEPS;
    rsum[t * NB + r] = r1[0];
  }
}

// ---------------------------------------------------------------------------
// 256x128 distance tile: S = Ah @ Bh^T via WMMA, fused distance + masked
// per-row max/min reduction -> global atomics. 8 waves, 64x64 per wave.
// ---------------------------------------------------------------------------
__global__ __launch_bounds__(256)
void tl_dist_tile_kernel(const _Float16* __restrict__ Ah,   // anchors  [NB][ND]
                         const _Float16* __restrict__ Bh,   // targets  [NB][ND]
                         const float*    __restrict__ na,   // ||A_i||^2
                         const float*    __restrict__ nbp,  // ||B_j+eps||^2
                         const float*    __restrict__ rsA,  // sum(A_i)
                         const int*      __restrict__ y,
                         unsigned*       __restrict__ smax, // per anchor row
                         unsigned*       __restrict__ dmin) {
  __shared__ _Float16 ldsA[2][BM * LROW];   // 2 x 20 KB
  __shared__ _Float16 ldsB[2][BN * LROW];   // 2 x 10 KB

  const int tid  = threadIdx.x;
  const int lane = tid & 31;
  const int w    = tid >> 5;      // wave 0..7
  const int wm   = w & 3;         // 4 waves along M: 64 rows each
  const int wn   = w >> 2;        // 2 waves along N: 64 cols each
  const int half = lane >> 4;     // 0: lanes 0-15, 1: lanes 16-31
  const int c16  = lane & 15;
  const int rowBase = blockIdx.y * BM;
  const int colBase = blockIdx.x * BN;

  v8f acc[4][4] = {};

  // stage one BK-slab (A: 256x32 halves = 1024 b128 chunks, B: 128x32 = 512)
  auto stage = [&](int kk, int buf) {
#if TL_ASYNC
#pragma unroll
    for (int p = 0; p < 4; ++p) {
      int chunk = tid + p * 256;
      int row = chunk >> 2, c8 = (chunk & 3) << 3;
      __builtin_amdgcn_global_load_async_to_lds_b128(
          (tl_gvp)(Ah + (size_t)(rowBase + row) * ND + kk + c8),
          (tl_lvp)&ldsA[buf][row * LROW + c8], 0, 0);
    }
#pragma unroll
    for (int p = 0; p < 2; ++p) {
      int chunk = tid + p * 256;
      int row = chunk >> 2, c8 = (chunk & 3) << 3;
      __builtin_amdgcn_global_load_async_to_lds_b128(
          (tl_gvp)(Bh + (size_t)(colBase + row) * ND + kk + c8),
          (tl_lvp)&ldsB[buf][row * LROW + c8], 0, 0);
    }
#else
#pragma unroll
    for (int p = 0; p < 4; ++p) {
      int chunk = tid + p * 256;
      int row = chunk >> 2, c8 = (chunk & 3) << 3;
      float4 va = *reinterpret_cast<const float4*>(
          Ah + (size_t)(rowBase + row) * ND + kk + c8);
      *reinterpret_cast<float4*>(&ldsA[buf][row * LROW + c8]) = va;
    }
#pragma unroll
    for (int p = 0; p < 2; ++p) {
      int chunk = tid + p * 256;
      int row = chunk >> 2, c8 = (chunk & 3) << 3;
      float4 vb = *reinterpret_cast<const float4*>(
          Bh + (size_t)(colBase + row) * ND + kk + c8);
      *reinterpret_cast<float4*>(&ldsB[buf][row * LROW + c8]) = vb;
    }
    __builtin_prefetch(Ah + (size_t)(rowBase + (tid >> 1)) * ND + kk + BK, 0, 1);
#endif
  };
  auto stage_wait = [&]() {
#if TL_ASYNC
    asm volatile("s_wait_asynccnt 0" ::: "memory");
#endif
  };

  stage(0, 0);
  stage_wait();
  __syncthreads();

  for (int kk = 0; kk < ND; kk += BK) {
    const int buf = (kk >> 5) & 1;
    if (kk + BK < ND) stage(kk + BK, buf ^ 1);   // fill other buffer

    // ---- fragments (ISA 16-bit 16x32 layout):
    // lanes 0-15: M=lane, K in {0..7}U{16..23}; lanes 16-31: M=lane-16, K+8
    const int c0 = half * 8;   // halves
    v16h afr[4], bfr[4];
#pragma unroll
    for (int mt = 0; mt < 4; ++mt) {
      int rr = wm * 64 + mt * 16 + c16;
      float4* ap = reinterpret_cast<float4*>(&afr[mt]);
      ap[0] = *reinterpret_cast<const float4*>(&ldsA[buf][rr * LROW + c0]);
      ap[1] = *reinterpret_cast<const float4*>(&ldsA[buf][rr * LROW + c0 + 16]);
    }
#pragma unroll
    for (int nt = 0; nt < 4; ++nt) {
      int rr = wn * 64 + nt * 16 + c16;
      float4* bp = reinterpret_cast<float4*>(&bfr[nt]);
      bp[0] = *reinterpret_cast<const float4*>(&ldsB[buf][rr * LROW + c0]);
      bp[1] = *reinterpret_cast<const float4*>(&ldsB[buf][rr * LROW + c0 + 16]);
    }

#pragma unroll
    for (int mt = 0; mt < 4; ++mt)
#pragma unroll
      for (int nt = 0; nt < 4; ++nt)
        acc[mt][nt] = __builtin_amdgcn_wmma_f32_16x16x32_f16(
            false, afr[mt], false, bfr[nt], (short)0, acc[mt][nt], false, false);

    if (kk + BK < ND) stage_wait();              // other buffer landed
    __syncthreads();                             // everyone done with `buf`
  }

  // ---- epilogue: distance + masked row max/min, lane reduce, atomics ----
  // C layout: VGPR v, lanes 0-15 -> M=v, lanes 16-31 -> M=v+8; N = lane&15.
#pragma unroll
  for (int mt = 0; mt < 4; ++mt) {
#pragma unroll
    for (int v = 0; v < 8; ++v) {
      const int i  = rowBase + wm * 64 + mt * 16 + half * 8 + v;
      const float nai = na[i];
      const float rsi = rsA[i];
      const int   yi  = y[i];

      float sv = 0.0f;
      float dv = 3.0e38f;
#pragma unroll
      for (int nt = 0; nt < 4; ++nt) {
        const int j = colBase + wn * 64 + nt * 16 + c16;
        float S  = acc[mt][nt][v];
        float sq = nai + nbp[j] - 2.0f * (S + FEPS * rsi);
        float d  = sqrtf(fmaxf(sq, 0.0f));
        bool same = (yi == y[j]);
        sv = fmaxf(sv, same ? d : 0.0f);
        dv = fminf(dv, same ? 3.0e38f : d);
      }
      // reduce across the 16 lanes holding this row (D>=0: IEEE order==uint order)
#pragma unroll
      for (int o = 1; o < 16; o <<= 1) {
        sv = fmaxf(sv, __shfl_xor(sv, o, 32));
        dv = fminf(dv, __shfl_xor(dv, o, 32));
      }
      if (c16 == 0) {
        atomicMax(&smax[i], __float_as_uint(sv));
        atomicMin(&dmin[i], __float_as_uint(dv));
      }
    }
  }
}

// ---------------------------------------------------------------------------
__global__ void tl_final_kernel(const unsigned* __restrict__ smax,
                                const unsigned* __restrict__ dmin,
                                float* __restrict__ out) {
  __shared__ float red[256];
  float acc = 0.f;
  for (int idx = threadIdx.x; idx < 2 * NB; idx += 256) {
    float s  = __uint_as_float(smax[idx]);
    float dm = __uint_as_float(dmin[idx]);
    float l  = s - dm + FXI;
    acc += (l > 0.f) ? l : 0.f;
  }
  red[threadIdx.x] = acc;
  __syncthreads();
  for (int o = 128; o > 0; o >>= 1) {
    if (threadIdx.x < (unsigned)o) red[threadIdx.x] += red[threadIdx.x + o];
    __syncthreads();
  }
  if (threadIdx.x == 0) out[0] = red[0];
}

// ---------------------------------------------------------------------------
extern "C" void kernel_launch(void* const* d_in, const int* in_sizes, int n_in,
                              void* d_out, int out_size, void* d_ws, size_t ws_size,
                              hipStream_t stream) {
  const float* v_f = (const float*)d_in[0];
  const float* i_f = (const float*)d_in[1];
  const int*   y   = (const int*)d_in[2];

  char* ws = (char*)d_ws;
  _Float16* h0 = (_Float16*)ws;                              // f16(i_f)  8 MB
  _Float16* h1 = (_Float16*)(ws + (size_t)NB * ND * 2);      // f16(v_f)  8 MB
  float* nsq   = (float*)(ws + (size_t)NB * ND * 4);         // [2][NB]
  float* neps  = nsq  + 2 * NB;                              // [2][NB]
  float* rsum  = neps + 2 * NB;                              // [2][NB]
  unsigned* smax = (unsigned*)(rsum + 2 * NB);               // [2][NB]
  unsigned* dmin = smax + 2 * NB;                            // [2][NB]

  tl_init_kernel<<<(2 * NB + 255) / 256, 256, 0, stream>>>(smax, dmin, 2 * NB);
  tl_prep_kernel<<<dim3(NB, 2), 256, 0, stream>>>(i_f, v_f, h0, h1, nsq, neps, rsum);

  dim3 grid(NB / BN, NB / BM);
  // direction 1: anchor = i_f, targets = v_f (+eps)
  tl_dist_tile_kernel<<<grid, 256, 0, stream>>>(h0, h1, nsq, neps + NB, rsum,
                                                y, smax, dmin);
  // direction 2: anchor = v_f, targets = i_f (+eps)
  tl_dist_tile_kernel<<<grid, 256, 0, stream>>>(h1, h0, nsq + NB, neps, rsum + NB,
                                                y, smax + NB, dmin + NB);

  tl_final_kernel<<<1, 256, 0, stream>>>(smax, dmin, (float*)d_out);
}
